// SparseMoE_11914239279741
// MI455X (gfx1250) — matrix operand order repaired and checked by reading, not compiled, not run on hardware
//
#include <hip/hip_runtime.h>
#include <cstdint>
#include <cstddef>

// ---------------- problem constants (from reference) ----------------
#define T_TOKENS 4096          // 2 * 2048 tokens
#define D_EMBD   1024
#define N_EXP    8
#define DFF      4096          // 4 * N_EMBD
#define TOPK     2
#define BK       64
#define LSTR     72            // padded LDS row stride (bf16 elems): 144B = 36 dwords
#define BM       128           // block tile M
#define BN       256           // block tile N (2 M-waves x 4 N-waves, 64x64 per wave)

// ---------------- WMMA types ----------------
typedef __bf16 bf16_t;
typedef bf16_t v16bf __attribute__((ext_vector_type(16)));
typedef bf16_t v2bf  __attribute__((ext_vector_type(2)));
typedef float  v8f   __attribute__((ext_vector_type(8)));

union FragAB { v16bf v; uint4 q[2]; };

// pack two fp32 -> packed bf16x2
__device__ inline unsigned bf16pack(float a, float b) {
#if __has_builtin(__builtin_amdgcn_cvt_pk_bf16_f32)
  union { v2bf v; unsigned u; } r;
  r.v = __builtin_amdgcn_cvt_pk_bf16_f32(a, b);
  return r.u;
#else
  unsigned ua = __float_as_uint(a), ub = __float_as_uint(b);
  ua = (ua + 0x7FFFu + ((ua >> 16) & 1u)) >> 16;
  ub = (ub + 0x7FFFu + ((ub >> 16) & 1u)) >> 16;
  return ua | (ub << 16);
#endif
}

// CDNA5 async global->LDS copy (tracked by ASYNCcnt)
__device__ inline void async_g2l_b128(unsigned lds_byte_addr, const void* gaddr) {
  asm volatile("global_load_async_to_lds_b128 %0, %1, off"
               :: "v"(lds_byte_addr), "v"(gaddr) : "memory");
}
__device__ inline void wait_async0() {
#if __has_builtin(__builtin_amdgcn_s_wait_asynccnt)
  __builtin_amdgcn_s_wait_asynccnt(0);
#else
  asm volatile("s_wait_asynccnt 0" ::: "memory");
#endif
}

// ---------------- workspace layout (bytes) ----------------
static constexpr size_t COUNTS_OFF  = 0;                                  // 8 ints
static constexpr size_t CURSORS_OFF = 32;                                 // 8 ints
static constexpr size_t OFFSETS_OFF = 64;                                 // 8 ints
static constexpr size_t TOPIDX_OFF  = 128;                                // T*2 ints
static constexpr size_t TOPP_OFF    = TOPIDX_OFF + (size_t)T_TOKENS*2*4;  // T*2 floats
static constexpr size_t ROWLIST_OFF = TOPP_OFF   + (size_t)T_TOKENS*2*4;  // 2T ints
static constexpr size_t ROWPROB_OFF = ROWLIST_OFF+ (size_t)T_TOKENS*2*4;  // 2T floats
static constexpr size_t H_OFF = ((ROWPROB_OFF + (size_t)T_TOKENS*2*4 + 255)/256)*256;
// h buffer: 2T rows x DFF bf16 = 67.1 MB

// ---------------- kernel 0: zero output + counters ----------------
__global__ __launch_bounds__(256) void moe_zero(float* __restrict__ out,
                                                int* __restrict__ counts,
                                                int* __restrict__ cursors) {
  int idx = blockIdx.x * 256 + threadIdx.x;
  float4 z = make_float4(0.f, 0.f, 0.f, 0.f);
  reinterpret_cast<float4*>(out)[idx] = z;
  if (idx < N_EXP) { counts[idx] = 0; cursors[idx] = 0; }
}

// ---------------- kernel 1: router (1 wave per token) ----------------
__global__ __launch_bounds__(32) void moe_router(const float* __restrict__ x,
                                                 const float* __restrict__ Wr,
                                                 const float* __restrict__ br,
                                                 int* __restrict__ counts,
                                                 int* __restrict__ top_idx,
                                                 float* __restrict__ top_p) {
  const int t = blockIdx.x;
  const int lane = threadIdx.x;
  const float* xr = x + (size_t)t * D_EMBD;
  float acc[N_EXP];
#pragma unroll
  for (int e = 0; e < N_EXP; ++e) acc[e] = 0.f;
  for (int j = lane; j < D_EMBD; j += 32) {
    float xv = xr[j];
    const float4* wr = reinterpret_cast<const float4*>(Wr + (size_t)j * N_EXP);
    float4 w0 = wr[0], w1 = wr[1];
    acc[0] += xv * w0.x; acc[1] += xv * w0.y; acc[2] += xv * w0.z; acc[3] += xv * w0.w;
    acc[4] += xv * w1.x; acc[5] += xv * w1.y; acc[6] += xv * w1.z; acc[7] += xv * w1.w;
  }
#pragma unroll
  for (int e = 0; e < N_EXP; ++e) {
#pragma unroll
    for (int s = 16; s > 0; s >>= 1) acc[e] += __shfl_down(acc[e], s, 32);
  }
  if (lane == 0) {
    float lg[N_EXP];
#pragma unroll
    for (int e = 0; e < N_EXP; ++e) lg[e] = acc[e] + br[e];
    int e1 = 0;
#pragma unroll
    for (int e = 1; e < N_EXP; ++e) if (lg[e] > lg[e1]) e1 = e;
    int e2 = (e1 == 0) ? 1 : 0;
#pragma unroll
    for (int e = 0; e < N_EXP; ++e) if (e != e1 && lg[e] > lg[e2]) e2 = e;
    float d  = expf(lg[e2] - lg[e1]);
    float p1 = 1.f / (1.f + d);
    float p2 = d / (1.f + d);
    top_idx[t*2+0] = e1; top_idx[t*2+1] = e2;
    top_p [t*2+0] = p1; top_p [t*2+1] = p2;
    atomicAdd(&counts[e1], 1);
    atomicAdd(&counts[e2], 1);
  }
}

// ---------------- kernel 2: tiny exclusive scan over 8 counts ----------------
__global__ void moe_scan(const int* __restrict__ counts,
                         int* __restrict__ offsets,
                         int* __restrict__ cursors) {
  if (threadIdx.x == 0) {
    int run = 0;
    for (int e = 0; e < N_EXP; ++e) { offsets[e] = run; run += counts[e]; cursors[e] = 0; }
  }
}

// ---------------- kernel 3: build per-expert token lists ----------------
__global__ __launch_bounds__(256) void moe_fill(const int* __restrict__ top_idx,
                                                const float* __restrict__ top_p,
                                                const int* __restrict__ offsets,
                                                int* __restrict__ cursors,
                                                int* __restrict__ rowlist,
                                                float* __restrict__ rowprob) {
  int t = blockIdx.x * 256 + threadIdx.x;
  if (t >= T_TOKENS) return;
#pragma unroll
  for (int j = 0; j < TOPK; ++j) {
    int e = top_idx[t*2 + j];
    float p = top_p[t*2 + j];
    int pos = atomicAdd(&cursors[e], 1);
    rowlist[offsets[e] + pos] = t;
    rowprob[offsets[e] + pos] = p;
  }
}

// ---------------- kernel 4: grouped GEMM1  h = relu(gather(x) @ W1[e] + b1[e]) ----------------
__global__ __launch_bounds__(256) void moe_gemm1(const float* __restrict__ x,
                                                 const float* __restrict__ W1,
                                                 const float* __restrict__ b1,
                                                 const int* __restrict__ counts,
                                                 const int* __restrict__ offsets,
                                                 const int* __restrict__ rowlist,
                                                 bf16_t* __restrict__ hbuf) {
  const int e    = blockIdx.z;
  const int cnt  = counts[e];
  const int row0 = blockIdx.x * BM;
  if (row0 >= cnt) return;
  const int n0  = blockIdx.y * BN;
  const int off = offsets[e];
  const float* W1e = W1 + (size_t)e * D_EMBD * DFF;

  __shared__ alignas(16) bf16_t ldsA[BM][LSTR];   // [row][k], padded rows
  __shared__ alignas(16) bf16_t ldsB[BN][LSTR];   // [n][k], padded rows

  const int tid  = threadIdx.x;
  const int lane = tid & 31;
  const int wave = tid >> 5;
  const int wm   = wave & 1;    // 2 waves over M (64 rows each)
  const int wn   = wave >> 1;   // 4 waves over N (64 cols each)

  // clamped gather indices (branch-free; padded rows predicated at epilogue)
  int tokc[8];
#pragma unroll
  for (int it = 0; it < 8; ++it) {
    int lin  = tid + it * 256;
    int slot = row0 + (lin >> 4);
    tokc[it] = rowlist[off + (slot < cnt ? slot : cnt - 1)];
  }

  v8f acc[4][4];
#pragma unroll
  for (int mi = 0; mi < 4; ++mi)
#pragma unroll
    for (int ni = 0; ni < 4; ++ni) acc[mi][ni] = {};

  for (int k0 = 0; k0 < D_EMBD; k0 += BK) {
    // --- stage A (gathered tokens), fp32 -> packed bf16, 8B LDS stores ---
#pragma unroll
    for (int it = 0; it < 8; ++it) {
      int lin = tid + it * 256;         // 128 rows x 16 float4
      int r   = lin >> 4;
      int c4  = lin & 15;
      float4 v = *reinterpret_cast<const float4*>(x + (size_t)tokc[it] * D_EMBD + k0 + c4 * 4);
      uint2 u;
      u.x = bf16pack(v.x, v.y);
      u.y = bf16pack(v.z, v.w);
      *reinterpret_cast<uint2*>(&ldsA[r][c4 * 4]) = u;
    }
    // --- stage B (W1 slice): read 2 k-rows, pack k-pairs, 4B LDS stores ---
#pragma unroll
    for (int it = 0; it < 8; ++it) {
      int lin = tid + it * 256;         // 32 k-pairs x 64 float4-of-n
      int kp  = lin >> 6;
      int n4  = lin & 63;
      const float* s0 = W1e + (size_t)(k0 + 2 * kp) * DFF + n0 + n4 * 4;
      float4 va = *reinterpret_cast<const float4*>(s0);
      float4 vb = *reinterpret_cast<const float4*>(s0 + DFF);
      int n = n4 * 4;
      *reinterpret_cast<unsigned*>(&ldsB[n+0][2*kp]) = bf16pack(va.x, vb.x);
      *reinterpret_cast<unsigned*>(&ldsB[n+1][2*kp]) = bf16pack(va.y, vb.y);
      *reinterpret_cast<unsigned*>(&ldsB[n+2][2*kp]) = bf16pack(va.z, vb.z);
      *reinterpret_cast<unsigned*>(&ldsB[n+3][2*kp]) = bf16pack(va.w, vb.w);
    }
    if (k0 + BK < D_EMBD)
      __builtin_prefetch(W1e + (size_t)(k0 + BK) * DFF + n0 + (tid & 255), 0, 1);
    __syncthreads();

#pragma unroll
    for (int ks = 0; ks < BK; ks += 32) {
      FragAB a[4], b[4];
      const int khA = (lane < 16) ? 0 : 8;
      const int khB = (lane < 16) ? 0 : 8;
#pragma unroll
      for (int mi = 0; mi < 4; ++mi) {
        int m = wm * 64 + mi * 16 + (lane & 15);
        a[mi].q[0] = *reinterpret_cast<const uint4*>(&ldsA[m][ks + khA]);
        a[mi].q[1] = *reinterpret_cast<const uint4*>(&ldsA[m][ks + khA + 16]);
      }
#pragma unroll
      for (int ni = 0; ni < 4; ++ni) {
        int n = wn * 64 + ni * 16 + (lane & 15);
        b[ni].q[0] = *reinterpret_cast<const uint4*>(&ldsB[n][ks + khB]);
        b[ni].q[1] = *reinterpret_cast<const uint4*>(&ldsB[n][ks + khB + 16]);
      }
#pragma unroll
      for (int mi = 0; mi < 4; ++mi)
#pragma unroll
        for (int ni = 0; ni < 4; ++ni)
          acc[mi][ni] = __builtin_amdgcn_wmma_f32_16x16x32_bf16(
              false, a[mi].v, false, b[ni].v, (short)0, acc[mi][ni], false, false);
    }
    __syncthreads();
  }

  // --- epilogue: bias + relu, store bf16 h ---
  const float* b1e = b1 + (size_t)e * DFF;
#pragma unroll
  for (int mi = 0; mi < 4; ++mi) {
#pragma unroll
    for (int i = 0; i < 8; ++i) {
      int m = wm * 64 + mi * 16 + ((lane < 16) ? i : i + 8);
      int slot = row0 + m;
      if (slot < cnt) {
#pragma unroll
        for (int ni = 0; ni < 4; ++ni) {
          int col = n0 + wn * 64 + ni * 16 + (lane & 15);
          float v = acc[mi][ni][i] + b1e[col];
          v = v > 0.f ? v : 0.f;
          hbuf[(size_t)(off + slot) * DFF + col] = (bf16_t)v;
        }
      }
    }
  }
}

// ---------------- kernel 5: grouped GEMM2  out[tok] += p * (h @ W2[e] + b2[e]) ----------------
__global__ __launch_bounds__(256) void moe_gemm2(const bf16_t* __restrict__ hbuf,
                                                 const float* __restrict__ W2,
                                                 const float* __restrict__ b2,
                                                 const int* __restrict__ counts,
                                                 const int* __restrict__ offsets,
                                                 const int* __restrict__ rowlist,
                                                 const float* __restrict__ rowprob,
                                                 float* __restrict__ out) {
  const int e    = blockIdx.z;
  const int cnt  = counts[e];
  const int row0 = blockIdx.x * BM;
  if (row0 >= cnt) return;
  const int n0  = blockIdx.y * BN;
  const int off = offsets[e];
  const float* W2e = W2 + (size_t)e * DFF * D_EMBD;

  __shared__ alignas(16) bf16_t ldsA[BM][LSTR];
  __shared__ alignas(16) bf16_t ldsB[BN][LSTR];

  const int tid  = threadIdx.x;
  const int lane = tid & 31;
  const int wave = tid >> 5;
  const int wm   = wave & 1;
  const int wn   = wave >> 1;

  // clamped h-row base pointers (branch-free async staging)
  const bf16_t* srcA[4];
#pragma unroll
  for (int it = 0; it < 4; ++it) {
    int lin  = tid + it * 256;
    int slot = row0 + (lin >> 3);
    srcA[it] = hbuf + (size_t)(off + (slot < cnt ? slot : cnt - 1)) * DFF;
  }

  v8f acc[4][4];
#pragma unroll
  for (int mi = 0; mi < 4; ++mi)
#pragma unroll
    for (int ni = 0; ni < 4; ++ni) acc[mi][ni] = {};

  for (int k0 = 0; k0 < DFF; k0 += BK) {
    // --- stage A (h rows, already bf16) via CDNA5 async global->LDS b128 ---
#pragma unroll
    for (int it = 0; it < 4; ++it) {
      int lin = tid + it * 256;        // 128 rows x 8 uint4
      int r   = lin >> 3;
      int c8  = lin & 7;
      unsigned lds_addr = (unsigned)(unsigned long long)(uintptr_t)&ldsA[r][c8 * 8];
      async_g2l_b128(lds_addr, srcA[it] + k0 + c8 * 8);
    }
    // --- stage B (W2 slice): read 2 k-rows, pack k-pairs, 4B LDS stores ---
#pragma unroll
    for (int it = 0; it < 8; ++it) {
      int lin = tid + it * 256;        // 32 k-pairs x 64 float4-of-n
      int kp  = lin >> 6;
      int n4  = lin & 63;
      const float* s0 = W2e + (size_t)(k0 + 2 * kp) * D_EMBD + n0 + n4 * 4;
      float4 va = *reinterpret_cast<const float4*>(s0);
      float4 vb = *reinterpret_cast<const float4*>(s0 + D_EMBD);
      int n = n4 * 4;
      *reinterpret_cast<unsigned*>(&ldsB[n+0][2*kp]) = bf16pack(va.x, vb.x);
      *reinterpret_cast<unsigned*>(&ldsB[n+1][2*kp]) = bf16pack(va.y, vb.y);
      *reinterpret_cast<unsigned*>(&ldsB[n+2][2*kp]) = bf16pack(va.z, vb.z);
      *reinterpret_cast<unsigned*>(&ldsB[n+3][2*kp]) = bf16pack(va.w, vb.w);
    }
    if (k0 + BK < DFF)
      __builtin_prefetch(W2e + (size_t)(k0 + BK) * D_EMBD + n0 + (tid & 255), 0, 1);
    wait_async0();
    __syncthreads();

#pragma unroll
    for (int ks = 0; ks < BK; ks += 32) {
      FragAB a[4], b[4];
      const int khA = (lane < 16) ? 0 : 8;
      const int khB = (lane < 16) ? 0 : 8;
#pragma unroll
      for (int mi = 0; mi < 4; ++mi) {
        int m = wm * 64 + mi * 16 + (lane & 15);
        a[mi].q[0] = *reinterpret_cast<const uint4*>(&ldsA[m][ks + khA]);
        a[mi].q[1] = *reinterpret_cast<const uint4*>(&ldsA[m][ks + khA + 16]);
      }
#pragma unroll
      for (int ni = 0; ni < 4; ++ni) {
        int n = wn * 64 + ni * 16 + (lane & 15);
        b[ni].q[0] = *reinterpret_cast<const uint4*>(&ldsB[n][ks + khB]);
        b[ni].q[1] = *reinterpret_cast<const uint4*>(&ldsB[n][ks + khB + 16]);
      }
#pragma unroll
      for (int mi = 0; mi < 4; ++mi)
#pragma unroll
        for (int ni = 0; ni < 4; ++ni)
          acc[mi][ni] = __builtin_amdgcn_wmma_f32_16x16x32_bf16(
              false, a[mi].v, false, b[ni].v, (short)0, acc[mi][ni], false, false);
    }
    __syncthreads();
  }

  // --- epilogue: bias, scale by routing prob, atomic scatter-add ---
  const float* b2e = b2 + (size_t)e * D_EMBD;
#pragma unroll
  for (int mi = 0; mi < 4; ++mi) {
#pragma unroll
    for (int i = 0; i < 8; ++i) {
      int m = wm * 64 + mi * 16 + ((lane < 16) ? i : i + 8);
      int slot = row0 + m;
      if (slot < cnt) {
        int tok = rowlist[off + slot];
        float p = rowprob[off + slot];
#pragma unroll
        for (int ni = 0; ni < 4; ++ni) {
          int col = n0 + wn * 64 + ni * 16 + (lane & 15);
          float v = acc[mi][ni][i] + b2e[col];
          atomicAdd(&out[(size_t)tok * D_EMBD + col], p * v);
        }
      }
    }
  }
}

// ---------------- launch ----------------
extern "C" void kernel_launch(void* const* d_in, const int* in_sizes, int n_in,
                              void* d_out, int out_size, void* d_ws, size_t ws_size,
                              hipStream_t stream) {
  const float* x  = (const float*)d_in[0];
  const float* Wr = (const float*)d_in[1];
  const float* br = (const float*)d_in[2];
  const float* W1 = (const float*)d_in[3];
  const float* b1 = (const float*)d_in[4];
  const float* W2 = (const float*)d_in[5];
  const float* b2 = (const float*)d_in[6];
  float* out = (float*)d_out;

  char* ws = (char*)d_ws;
  int*    counts  = (int*)  (ws + COUNTS_OFF);
  int*    cursors = (int*)  (ws + CURSORS_OFF);
  int*    offsets = (int*)  (ws + OFFSETS_OFF);
  int*    topidx  = (int*)  (ws + TOPIDX_OFF);
  float*  topp    = (float*)(ws + TOPP_OFF);
  int*    rowlist = (int*)  (ws + ROWLIST_OFF);
  float*  rowprob = (float*)(ws + ROWPROB_OFF);
  bf16_t* hbuf    = (bf16_t*)(ws + H_OFF);

  moe_zero  <<<4096, 256, 0, stream>>>(out, counts, cursors);
  moe_router<<<T_TOKENS, 32, 0, stream>>>(x, Wr, br, counts, topidx, topp);
  moe_scan  <<<1, 32, 0, stream>>>(counts, offsets, cursors);
  moe_fill  <<<T_TOKENS / 256, 256, 0, stream>>>(topidx, topp, offsets, cursors, rowlist, rowprob);
  moe_gemm1 <<<dim3(T_TOKENS / BM, DFF / BN, N_EXP), 256, 0, stream>>>(
      x, W1, b1, counts, offsets, rowlist, hbuf);
  moe_gemm2 <<<dim3(T_TOKENS / BM, D_EMBD / BN, N_EXP), 256, 0, stream>>>(
      hbuf, W2, b2, counts, offsets, rowlist, rowprob, out);
}
